// ReLUSAE_68058051772489
// MI455X (gfx1250) — matrix operand order repaired and checked by reading, not compiled, not run on hardware
//
#include <hip/hip_runtime.h>
#include <hip/hip_bf16.h>
#include <stdint.h>

#define N_TOK 16384
#define D_IN  1024
#define D_SAE 16384
#define K_TOP 64

#define KSTEP   64                 // K per pipeline stage
#define NSTAGE  (D_IN / KSTEP)     // 16 stages

typedef __attribute__((ext_vector_type(16))) __bf16 v16bf;
typedef __attribute__((ext_vector_type(8)))  __bf16 v8bf;
typedef __attribute__((ext_vector_type(8)))  float  v8f;

// ---------------------------------------------------------------------------
// CDNA5 async global->LDS copy (ASYNCcnt-tracked), CDNA5 ISA
// ---------------------------------------------------------------------------
__device__ __forceinline__ void async_copy_b128(unsigned lds_off, const void* gptr) {
  asm volatile("global_load_async_to_lds_b128 %0, %1, off"
               :: "v"(lds_off), "v"((unsigned long long)(uintptr_t)gptr)
               : "memory");
}
__device__ __forceinline__ void wait_async0() {
#if __has_builtin(__builtin_amdgcn_s_wait_asynccnt)
  __builtin_amdgcn_s_wait_asynccnt(0);
#else
  asm volatile("s_wait_asynccnt 0" ::: "memory");
#endif
}
__device__ __forceinline__ void wait_async8() {   // allow 8 newer ops in flight
#if __has_builtin(__builtin_amdgcn_s_wait_asynccnt)
  __builtin_amdgcn_s_wait_asynccnt(8);
#else
  asm volatile("s_wait_asynccnt 8" ::: "memory");
#endif
}

// ---------------------------------------------------------------------------
// Kernel 1: Xbf = bf16(x - b_dec)   [N_TOK, D_IN]
// ---------------------------------------------------------------------------
__global__ __launch_bounds__(256) void cvt_x_kernel(
    const float* __restrict__ x, const float* __restrict__ b_dec,
    unsigned short* __restrict__ Xbf)
{
  const size_t i = ((size_t)blockIdx.x * 256 + threadIdx.x) * 4;
  const float4 v = *(const float4*)(x + i);
  const unsigned c = (unsigned)(i & (D_IN - 1));
  const float4 b = *(const float4*)(b_dec + c);
  __bf16* o = (__bf16*)(Xbf + i);
  o[0] = (__bf16)(v.x - b.x);
  o[1] = (__bf16)(v.y - b.y);
  o[2] = (__bf16)(v.z - b.z);
  o[3] = (__bf16)(v.w - b.w);
}

// ---------------------------------------------------------------------------
// Kernel 2: Wbf[n][k] = bf16(W_enc[k][n])  (transpose via 32x32 LDS tile)
// ---------------------------------------------------------------------------
__global__ __launch_bounds__(256) void cvt_w_kernel(
    const float* __restrict__ W_enc, unsigned short* __restrict__ Wbf)
{
  __shared__ float t[32][33];
  const int nt = blockIdx.x & (D_SAE / 32 - 1);   // 512 n-tiles
  const int kt = blockIdx.x >> 9;                 // 32  k-tiles
  const int lx = threadIdx.x & 31;
  const int ly = threadIdx.x >> 5;                // 0..7
#pragma unroll
  for (int j = 0; j < 4; ++j)
    t[ly + 8 * j][lx] =
        W_enc[(size_t)(kt * 32 + ly + 8 * j) * D_SAE + nt * 32 + lx];
  __syncthreads();
  __bf16* o = (__bf16*)Wbf;
#pragma unroll
  for (int j = 0; j < 4; ++j)
    o[(size_t)(nt * 32 + ly + 8 * j) * D_IN + kt * 32 + lx] =
        (__bf16)t[lx][ly + 8 * j];
}

// ---------------------------------------------------------------------------
// Kernel 3: encode GEMM  Z = relu(Xbf @ Wbf^T + b_enc)
// Block = 256 threads = 8 waves -> 128(M) x 128(N) macro tile.
// Wave = 64(M) x 32(N) = 4x2 WMMA 16x16x32 bf16 tiles, 2 K-subs per stage.
// Double-buffered LDS fed by async global->LDS DMA: stage s+1 is issued
// before computing stage s; s_wait_asynccnt 8 retires only the older stage.
// ---------------------------------------------------------------------------
__global__ __launch_bounds__(256) void encode_wmma_kernel(
    const unsigned short* __restrict__ Xbf,   // [N_TOK][D_IN] bf16
    const unsigned short* __restrict__ Wbf,   // [D_SAE][D_IN] bf16 (= W_enc^T)
    const float* __restrict__ b_enc,
    float* __restrict__ Z)                    // [N_TOK][D_SAE]
{
  __shared__ __bf16 sA[2 * 128 * KSTEP];      // 2 x 16 KB
  __shared__ __bf16 sB[2 * 128 * KSTEP];      // 2 x 16 KB

  const int tid  = threadIdx.x;
  const int lane = tid & 31;
  const int wid  = tid >> 5;
  const int wm   = wid & 1;          // 2 wave-rows of 64 M
  const int wn   = wid >> 1;         // 4 wave-cols of 32 N
  const int m0   = (blockIdx.x & 127) << 7;
  const int n0   = (blockIdx.x >> 7) << 7;

  const int half = lane >> 4;        // 0 | 1 (half-wave)
  const int l16  = lane & 15;

  v8f acc[4][2] = {};

  const unsigned ldsA = (unsigned)(size_t)&sA[0];
  const unsigned ldsB = (unsigned)(size_t)&sB[0];
  const __bf16* gX = (const __bf16*)Xbf;
  const __bf16* gW = (const __bf16*)Wbf;

  // Per-stage staging: 128 rows x 128 B for A and B = 1024 x 16B chunks each,
  // 4 chunks per matrix per thread (8 async ops / thread / stage).
  const int r  = (tid * 4) >> 3;            // not used; explicit below
  (void)r;

#define ISSUE_STAGE(k0, buf)                                                   \
  {                                                                            \
    _Pragma("unroll")                                                          \
    for (int i = 0; i < 4; ++i) {                                              \
      const int t  = tid + i * 256;          /* 0..1023 */                     \
      const int rr = t >> 3;                 /* 0..127  */                     \
      const int ch = t & 7;                  /* 16B chunk in 128B row */       \
      async_copy_b128(ldsA + (buf) * (128 * KSTEP * 2) + rr * 128 + ch * 16,   \
                      gX + (size_t)(m0 + rr) * D_IN + (k0) + ch * 8);          \
      async_copy_b128(ldsB + (buf) * (128 * KSTEP * 2) + rr * 128 + ch * 16,   \
                      gW + (size_t)(n0 + rr) * D_IN + (k0) + ch * 8);          \
    }                                                                          \
  }

  ISSUE_STAGE(0, 0)

  for (int st = 0; st < NSTAGE; ++st) {
    const int cur = st & 1;
    if (st + 1 < NSTAGE) {
      ISSUE_STAGE((st + 1) * KSTEP, cur ^ 1)
      wait_async8();                 // retire only stage st's 8 ops
    } else {
      wait_async0();
    }
    __syncthreads();

    const __bf16* bufA = sA + cur * (128 * KSTEP);
    const __bf16* bufB = sB + cur * (128 * KSTEP);

#pragma unroll
    for (int s = 0; s < 2; ++s) {            // two K=32 sub-steps
      v16bf af[4];
#pragma unroll
      for (int tm = 0; tm < 4; ++tm) {
        const int rr = wm * 64 + tm * 16 + l16;
        af[tm] = *(const v16bf*)(bufA + rr * KSTEP + s * 32 + half * 16);
      }
      v16bf bfr[2];
#pragma unroll
      for (int tn = 0; tn < 2; ++tn) {
        const int rr = wn * 32 + tn * 16 + l16;
        bfr[tn] = *(const v16bf*)(bufB + rr * KSTEP + s * 32 + half * 16);
      }
#pragma unroll
      for (int tm = 0; tm < 4; ++tm)
#pragma unroll
        for (int tn = 0; tn < 2; ++tn)
          acc[tm][tn] = __builtin_amdgcn_wmma_f32_16x16x32_bf16(
              false, af[tm], false, bfr[tn], (short)0, acc[tm][tn], false, false);
    }
    __syncthreads();                 // protect buffer reused by stage st+2
  }
#undef ISSUE_STAGE

  // Epilogue: C/D layout -> M = r + 8*half, N = l16. Fuse +b_enc and ReLU.
  // Non-temporal: 1 GB stream must not evict L2-resident weights.
#pragma unroll
  for (int tn = 0; tn < 2; ++tn) {
    const int col = n0 + wn * 32 + tn * 16 + l16;
    const float be = b_enc[col];
#pragma unroll
    for (int tm = 0; tm < 4; ++tm) {
#pragma unroll
      for (int rr = 0; rr < 8; ++rr) {
        const int grow = m0 + wm * 64 + tm * 16 + half * 8 + rr;
        float v = acc[tm][tn][rr] + be;
        v = (v > 0.0f) ? v : 0.0f;
        __builtin_nontemporal_store(v, &Z[(size_t)grow * D_SAE + col]);
      }
    }
  }
}

// ---------------------------------------------------------------------------
// Kernel 4: exact per-row top-K via 4-pass radix select on float bits
// (values >= 0 after ReLU -> bit pattern is order-preserving).
// The 64 KB row is staged once into LDS with async DMA; all histogram
// passes and the final masked write-back run out of LDS.
// ---------------------------------------------------------------------------
__global__ __launch_bounds__(256) void topk_kernel(
    float* __restrict__ Z, int* __restrict__ IDX, float* __restrict__ VAL)
{
  __shared__ float    srow[D_SAE];     // 64 KB
  __shared__ unsigned hist[256];
  __shared__ unsigned s_pref, s_kneed;
  __shared__ int s_tie, s_slot;

  const int row = blockIdx.x;
  const int tid = threadIdx.x;
  float* z = Z + (size_t)row * D_SAE;

  // Stage the row: 4096 x 16B chunks, 16 per thread, ASYNCcnt-tracked.
  const unsigned lds0 = (unsigned)(size_t)&srow[0];
#pragma unroll
  for (int i = 0; i < 16; ++i) {
    const int ch = tid * 16 + i;               // 0..4095
    async_copy_b128(lds0 + ch * 16, z + ch * 4);
  }
  wait_async0();
  __syncthreads();

  unsigned prefix = 0;
  int kneed = K_TOP;

  for (int shift = 24; shift >= 0; shift -= 8) {
    hist[tid] = 0;
    __syncthreads();
    const unsigned himask = (shift == 24) ? 0u : (0xFFFFFFFFu << (shift + 8));
#pragma unroll 4
    for (int j = 0; j < D_SAE / 256; ++j) {
      const unsigned v = __float_as_uint(srow[tid + j * 256]);
      if ((v & himask) == prefix)
        atomicAdd(&hist[(v >> shift) & 255u], 1u);
    }
    __syncthreads();
    if (tid == 0) {
      int cum = 0;
      int b = 255;
      for (; b > 0; --b) {
        const int h = (int)hist[b];
        if (cum + h >= kneed) break;
        cum += h;
      }
      s_pref  = prefix | ((unsigned)b << shift);
      s_kneed = (unsigned)(kneed - cum);
    }
    __syncthreads();
    prefix = s_pref;
    kneed  = (int)s_kneed;
  }

  if (tid == 0) { s_tie = 0; s_slot = 0; }
  __syncthreads();

  const unsigned thrbits = prefix;             // K-th largest bit pattern
  const float thr = __uint_as_float(thrbits);
  for (int j = 0; j < D_SAE / 256; ++j) {
    const int col = tid + j * 256;
    const float v = srow[col];
    bool keep;
    if (v > thr)                            keep = true;
    else if (__float_as_uint(v) == thrbits) keep = (atomicAdd(&s_tie, 1) < kneed);
    else                                    keep = false;
    if (keep) {
      const int slot = atomicAdd(&s_slot, 1);
      IDX[row * K_TOP + slot] = col;
      VAL[row * K_TOP + slot] = v;
      // z[col] already holds v; no write needed
    } else {
      __builtin_nontemporal_store(0.0f, &z[col]);
    }
  }
}

// ---------------------------------------------------------------------------
// Kernel 5: sparse decode  x_hat[row] = b_dec + sum_t val[t] * W_dec[idx[t], :]
// W_dec (64 MB) is L2-resident; ~2 GFLOP so fp32 VALU is fine.
// ---------------------------------------------------------------------------
__global__ __launch_bounds__(256) void decode_kernel(
    const int* __restrict__ IDX, const float* __restrict__ VAL,
    const float* __restrict__ W_dec, const float* __restrict__ b_dec,
    float* __restrict__ Xhat)
{
  const int row = blockIdx.x;
  const int tid = threadIdx.x;
  __shared__ int   sidx[K_TOP];
  __shared__ float sval[K_TOP];
  if (tid < K_TOP) {
    sidx[tid] = IDX[row * K_TOP + tid];
    sval[tid] = VAL[row * K_TOP + tid];
  }
  __syncthreads();

  const int c0 = tid * 4;                      // 256 threads x 4 cols = 1024
  float4 acc = *(const float4*)(b_dec + c0);
#pragma unroll 8
  for (int t = 0; t < K_TOP; ++t) {
    const float  v = sval[t];
    const float4 w = *(const float4*)(W_dec + (size_t)sidx[t] * D_IN + c0);
    acc.x = fmaf(v, w.x, acc.x);
    acc.y = fmaf(v, w.y, acc.y);
    acc.z = fmaf(v, w.z, acc.z);
    acc.w = fmaf(v, w.w, acc.w);
  }
  *(float4*)(Xhat + (size_t)row * D_IN + c0) = acc;
}

// ---------------------------------------------------------------------------
// Host launcher
// ---------------------------------------------------------------------------
extern "C" void kernel_launch(void* const* d_in, const int* in_sizes, int n_in,
                              void* d_out, int out_size, void* d_ws, size_t ws_size,
                              hipStream_t stream)
{
  (void)in_sizes; (void)n_in; (void)out_size; (void)ws_size;

  const float* x     = (const float*)d_in[0];
  const float* W_enc = (const float*)d_in[1];
  const float* b_enc = (const float*)d_in[2];
  const float* W_dec = (const float*)d_in[3];
  const float* b_dec = (const float*)d_in[4];
  // d_in[5] = k (always 64; tiled at compile time)

  float* xhat = (float*)d_out;                               // [N_TOK, D_IN]
  float* z    = (float*)d_out + (size_t)N_TOK * D_IN;        // [N_TOK, D_SAE]

  char* ws = (char*)d_ws;
  unsigned short* Xbf = (unsigned short*)ws;                                   // 32 MB
  unsigned short* Wbf = (unsigned short*)(ws + (size_t)N_TOK * D_IN * 2);      // 32 MB
  int*   IDX = (int*)(ws + (size_t)N_TOK * D_IN * 2 + (size_t)D_SAE * D_IN * 2);
  float* VAL = (float*)((char*)IDX + (size_t)N_TOK * K_TOP * 4);

  hipLaunchKernelGGL(cvt_x_kernel, dim3(N_TOK * D_IN / 4 / 256), dim3(256), 0, stream,
                     x, b_dec, Xbf);
  hipLaunchKernelGGL(cvt_w_kernel, dim3((D_SAE / 32) * (D_IN / 32)), dim3(256), 0, stream,
                     W_enc, Wbf);
  hipLaunchKernelGGL(encode_wmma_kernel, dim3((N_TOK / 128) * (D_SAE / 128)), dim3(256), 0, stream,
                     Xbf, Wbf, b_enc, z);
  hipLaunchKernelGGL(topk_kernel, dim3(N_TOK), dim3(256), 0, stream,
                     z, IDX, VAL);
  hipLaunchKernelGGL(decode_kernel, dim3(N_TOK), dim3(256), 0, stream,
                     IDX, VAL, W_dec, b_dec, xhat);
}
